// CausalSelfAttention_71940702208630
// MI455X (gfx1250) — matrix operand order repaired and checked by reading, not compiled, not run on hardware
//
#include <hip/hip_runtime.h>

// ---------------------------------------------------------------------------
// CausalSelfAttention forward for MI455X (gfx1250), bf16 WMMA pipeline.
// B=8, T=1024, C=512, H=8, D=64.  Flash-attention streaming with async
// Global->LDS double-buffered staging of K/V blocks (ASYNCcnt path).
// ---------------------------------------------------------------------------

#define BB 8
#define TT 1024
#define CC 512
#define HH 8
#define DD 64
#define MTOT (BB * TT)   // 8192

typedef __attribute__((ext_vector_type(16))) __bf16 v16bf;
typedef __attribute__((ext_vector_type(8)))  float  v8f;

union Frag32 {            // 32 bytes = one WMMA bf16 A/B fragment (8 VGPRs)
    v16bf v;
    uint4 u[2];
};

static __device__ __forceinline__ unsigned short f2bf(float f) {
    unsigned u = __float_as_uint(f);
    u += 0x7FFFu + ((u >> 16) & 1u);   // round-to-nearest-even
    return (unsigned short)(u >> 16);
}

static __device__ __forceinline__ v8f wmma_bf16(const v16bf& a, const v16bf& b, const v8f& c) {
    return __builtin_amdgcn_wmma_f32_16x16x32_bf16(false, a, false, b, (short)0, c,
                                                   false, false);
}

// ---- async Global->LDS (CDNA5 ASYNCcnt path) via inline asm ---------------
// GLOBAL_LOAD_ASYNC_TO_LDS_B128 (VGLOBAL op 98): VDST = per-lane LDS byte
// address, VADDR = 64-bit global address (GV mode, SADDR=off).  Tracked by
// ASYNCcnt; completion enforced with s_wait_asynccnt before the barrier.
static __device__ __forceinline__ void async_cp16(const unsigned short* g,
                                                  unsigned short* l) {
#if defined(__gfx1250__)
    unsigned lds_off = (unsigned)(uintptr_t)l;          // low 32 bits = AS3 addr
    unsigned long long ga = (unsigned long long)(uintptr_t)g;
    asm volatile("global_load_async_to_lds_b128 %0, %1, off"
                 :: "v"(lds_off), "v"(ga)
                 : "memory");
#else
    *(uint4*)l = *(const uint4*)g;
#endif
}

static __device__ __forceinline__ void wait_async0() {
#if defined(__gfx1250__)
    asm volatile("s_wait_asynccnt 0x0" ::: "memory");
#endif
}

// ---------------------------------------------------------------------------
// Stage 0: fp32 -> bf16 conversion
// ---------------------------------------------------------------------------
__global__ void cvt_f32_to_bf16(const float* __restrict__ src,
                                unsigned short* __restrict__ dst, int n) {
    int i = blockIdx.x * blockDim.x + threadIdx.x;
    if (i < n) dst[i] = f2bf(src[i]);
}

// ---------------------------------------------------------------------------
// Stage 1: fused QKV projection.  q/k/v[m][n] = sum_k x[m][k] * W[n][k] + b[n]
// Q,K stored as (b,h,t,d) bf16; V stored transposed as (b,h,d,t) bf16.
// 128 threads = 4 waves; 128x64 workgroup tile; wave = 64x32 (4x2 WMMA).
// ---------------------------------------------------------------------------
__global__ __launch_bounds__(128) void qkv_gemm_kernel(
    const unsigned short* __restrict__ xb,   // 8192 x 512
    const unsigned short* __restrict__ wqb,  // 512 x 512  [n][k]
    const unsigned short* __restrict__ wkb,
    const unsigned short* __restrict__ wvb,
    const float* __restrict__ bq,
    const float* __restrict__ bk,
    const float* __restrict__ bv,
    unsigned short* __restrict__ Qb,         // (b,h,t,d)
    unsigned short* __restrict__ Kb,         // (b,h,t,d)
    unsigned short* __restrict__ Vt)         // (b,h,d,t)
{
    const int lane = threadIdx.x & 31;
    const int wave = threadIdx.x >> 5;
    const int lm = lane & 15;
    const int lh = lane >> 4;
    const int wm = wave & 1, wn = wave >> 1;
    const int m0 = blockIdx.x * 128 + wm * 64;
    const int n0 = blockIdx.y * 64 + wn * 32;
    const int z  = blockIdx.z;       // 0=Q 1=K 2=V

    const unsigned short* W    = (z == 0) ? wqb : (z == 1) ? wkb : wvb;
    const float*          bias = (z == 0) ? bq  : (z == 1) ? bk  : bv;

    const v8f zero8 = {0.f, 0.f, 0.f, 0.f, 0.f, 0.f, 0.f, 0.f};
    v8f acc[4][2];
#pragma unroll
    for (int i = 0; i < 4; ++i)
#pragma unroll
        for (int j = 0; j < 2; ++j) acc[i][j] = zero8;

    for (int k0 = 0; k0 < CC; k0 += 32) {
        Frag32 a[4], bfr[2];
#pragma unroll
        for (int mt = 0; mt < 4; ++mt) {
            const unsigned short* p = xb + (size_t)(m0 + mt * 16 + lm) * CC + k0;
            a[mt].u[0] = *(const uint4*)(p + lh * 16);       // K = lh*16 .. +7
            a[mt].u[1] = *(const uint4*)(p + lh * 16 + 8);   // K = lh*16+8 .. +15
        }
#pragma unroll
        for (int nt = 0; nt < 2; ++nt) {
            const unsigned short* p = W + (size_t)(n0 + nt * 16 + lm) * CC + k0 + lh * 16;
            bfr[nt].u[0] = *(const uint4*)(p);               // K = lh*16 .. +7
            bfr[nt].u[1] = *(const uint4*)(p + 8);           // K = lh*16+8 .. +15
        }
#pragma unroll
        for (int mt = 0; mt < 4; ++mt)
#pragma unroll
            for (int nt = 0; nt < 2; ++nt)
                acc[mt][nt] = wmma_bf16(a[mt].v, bfr[nt].v, acc[mt][nt]);
    }

    // Epilogue: bias add, bf16 convert, layout-specific scatter.
#pragma unroll
    for (int mt = 0; mt < 4; ++mt) {
#pragma unroll
        for (int nt = 0; nt < 2; ++nt) {
            const int n  = n0 + nt * 16 + lm;
            const float bv_ = bias[n];
            const int h = n >> 6, d = n & 63;
#pragma unroll
            for (int r = 0; r < 8; ++r) {
                const int m = m0 + mt * 16 + r + 8 * lh;
                const int b = m >> 10, t = m & 1023;
                const unsigned short val = f2bf(acc[mt][nt][r] + bv_);
                if (z == 2)
                    Vt[(((size_t)(b * HH + h) * DD + d) * TT) + t] = val;
                else if (z == 1)
                    Kb[(((size_t)(b * HH + h) * TT + t) * DD) + d] = val;
                else
                    Qb[(((size_t)(b * HH + h) * TT + t) * DD) + d] = val;
            }
        }
    }
}

// ---------------------------------------------------------------------------
// Stage 2: flash attention.  Grid: (T/128, B*H).  256 threads = 8 waves,
// wave w owns q-rows [qb*128 + 16w, +16).  K/V blocks (64 wide) are staged
// into double-buffered LDS via async copies issued one block ahead, so the
// whole S/softmax/PV phase hides the copy latency.  LDS rows padded to 88
// elements (176B): 16B-aligned, conflict-free b128 fragment reads.
// ---------------------------------------------------------------------------
#define LPAD 88

static __device__ __forceinline__ void stage_kv(
    const unsigned short* __restrict__ Kh,
    const unsigned short* __restrict__ Vh,
    int kbase, int tid,
    unsigned short (*Kl)[LPAD], unsigned short (*Vl)[LPAD]) {
#pragma unroll
    for (int i = 0; i < 2; ++i) {
        const int c = tid + i * 256;          // 0..511
        const int r = c >> 3, p = c & 7;      // row, 16B-chunk
        async_cp16(Kh + (size_t)(kbase + r) * DD + p * 8, &Kl[r][p * 8]);
        async_cp16(Vh + (size_t)r * TT + kbase + p * 8,   &Vl[r][p * 8]);
    }
}

__global__ __launch_bounds__(256) void attn_kernel(
    const unsigned short* __restrict__ Qb,   // (b,h,t,d)
    const unsigned short* __restrict__ Kb,   // (b,h,t,d)
    const unsigned short* __restrict__ Vt,   // (b,h,d,t)
    unsigned short* __restrict__ yb)         // 8192 x 512, col = h*64+d
{
    __shared__ __align__(16) unsigned short Klds[2][64][LPAD];
    __shared__ __align__(16) unsigned short Vlds[2][64][LPAD];
    __shared__ __align__(16) unsigned short Plds[8][16][LPAD];

    const int tid  = threadIdx.x;
    const int lane = tid & 31;
    const int wave = tid >> 5;
    const int lm = lane & 15;
    const int lh = lane >> 4;
    const int qb = blockIdx.x;                // q-block 0..7 (128 rows each)
    const int bh = blockIdx.y;                // 0..63
    const int rowbase = qb * 128 + wave * 16; // wave's first q row
    const int jmax = 2 * qb + 1;              // last 64-wide k-block needed

    const unsigned short* Qh = Qb + (size_t)bh * TT * DD;
    const unsigned short* Kh = Kb + (size_t)bh * TT * DD;
    const unsigned short* Vh = Vt + (size_t)bh * DD * TT;  // row d, stride T

    // Q A-fragments: frag f covers head-dims d = f*32 .. f*32+31
    Frag32 qa[2];
#pragma unroll
    for (int f = 0; f < 2; ++f) {
        const unsigned short* p = Qh + (size_t)(rowbase + lm) * DD + f * 32;
        qa[f].u[0] = *(const uint4*)(p + lh * 16);
        qa[f].u[1] = *(const uint4*)(p + lh * 16 + 8);
    }

    const v8f zero8 = {0.f, 0.f, 0.f, 0.f, 0.f, 0.f, 0.f, 0.f};
    v8f o[4];
#pragma unroll
    for (int dt = 0; dt < 4; ++dt) o[dt] = zero8;

    float mrow[8], lrow[8];
#pragma unroll
    for (int r = 0; r < 8; ++r) { mrow[r] = -1e30f; lrow[r] = 0.f; }

    // stage K/V block 0 into buffer 0
    stage_kv(Kh, Vh, 0, tid, Klds[0], Vlds[0]);

    for (int j = 0; j <= jmax; ++j) {
        const int kbase = j * 64;
        const int buf = j & 1;

        wait_async0();     // this wave's copies for block j done
        __syncthreads();   // all waves' copies done; nobody reads buf^1 anymore

        // issue next block's copies into the other buffer; the whole
        // S/softmax/PV phase below hides their latency.
        if (j < jmax)
            stage_kv(Kh, Vh, (j + 1) * 64, tid, Klds[buf ^ 1], Vlds[buf ^ 1]);

        // ---- S = scale * Q Kᵀ (16 x 64 per wave), K from LDS ----
        v8f s[4];
#pragma unroll
        for (int nt = 0; nt < 4; ++nt) {
            v8f st = zero8;
#pragma unroll
            for (int f = 0; f < 2; ++f) {
                Frag32 kf;
                const unsigned short* p = &Klds[buf][nt * 16 + lm][f * 32 + lh * 16];
                kf.u[0] = *(const uint4*)(p);
                kf.u[1] = *(const uint4*)(p + 8);
                st = wmma_bf16(qa[f].v, kf.v, st);
            }
            s[nt] = st;
        }

        // ---- mask + scale + new row max ----
        float mnew[8];
#pragma unroll
        for (int r = 0; r < 8; ++r) mnew[r] = -1e30f;
#pragma unroll
        for (int nt = 0; nt < 4; ++nt) {
            const int col = kbase + nt * 16 + lm;
            const bool colok = (col % 25) != 24;
#pragma unroll
            for (int r = 0; r < 8; ++r) {
                const int row = rowbase + r + 8 * lh;
                float v = s[nt][r] * 0.125f;             // 1/sqrt(64)
                v = (colok && col <= row) ? v : -1e30f;
                s[nt][r] = v;
                mnew[r] = fmaxf(mnew[r], v);
            }
        }
#pragma unroll
        for (int r = 0; r < 8; ++r) {
#pragma unroll
            for (int off = 8; off >= 1; off >>= 1)
                mnew[r] = fmaxf(mnew[r], __shfl_xor(mnew[r], off, 32));
        }

        float alpha[8];
#pragma unroll
        for (int r = 0; r < 8; ++r) {
            const float mi = fmaxf(mrow[r], mnew[r]);
            alpha[r] = __expf(mrow[r] - mi);
            mrow[r] = mi;
        }

        // ---- P = exp(S - m), row sums ----
        float rsum[8];
#pragma unroll
        for (int r = 0; r < 8; ++r) rsum[r] = 0.f;
#pragma unroll
        for (int nt = 0; nt < 4; ++nt)
#pragma unroll
            for (int r = 0; r < 8; ++r) {
                const float p = __expf(s[nt][r] - mrow[r]);
                s[nt][r] = p;
                rsum[r] += p;
            }
#pragma unroll
        for (int r = 0; r < 8; ++r) {
#pragma unroll
            for (int off = 8; off >= 1; off >>= 1)
                rsum[r] += __shfl_xor(rsum[r], off, 32);
            lrow[r] = lrow[r] * alpha[r] + rsum[r];
        }

        // ---- rescale O, stage P into LDS (bf16) ----
#pragma unroll
        for (int dt = 0; dt < 4; ++dt)
#pragma unroll
            for (int r = 0; r < 8; ++r) o[dt][r] *= alpha[r];

#pragma unroll
        for (int nt = 0; nt < 4; ++nt)
#pragma unroll
            for (int r = 0; r < 8; ++r)
                Plds[wave][r + 8 * lh][nt * 16 + lm] = f2bf(s[nt][r]);

        __syncthreads();   // fences ushort-store -> b128-load of P

        // ---- O += P V  (A = P from LDS, B = Vᵀ rows from LDS) ----
        const unsigned short* prow = &Plds[wave][lm][0];
#pragma unroll
        for (int f = 0; f < 2; ++f) {            // t-chunk f*32 .. +31
            Frag32 pa;
            pa.u[0] = *(const uint4*)(prow + f * 32 + lh * 16);
            pa.u[1] = *(const uint4*)(prow + f * 32 + lh * 16 + 8);
#pragma unroll
            for (int dt = 0; dt < 4; ++dt) {
                Frag32 vf;
                const unsigned short* p = &Vlds[buf][dt * 16 + lm][f * 32 + lh * 16];
                vf.u[0] = *(const uint4*)(p);
                vf.u[1] = *(const uint4*)(p + 8);
                o[dt] = wmma_bf16(pa.v, vf.v, o[dt]);
            }
        }
    }

    // ---- normalize and store y (bf16, row-major 8192 x 512) ----
    const int b = bh >> 3, h = bh & 7;
#pragma unroll
    for (int dt = 0; dt < 4; ++dt) {
        const int d = dt * 16 + lm;
#pragma unroll
        for (int r = 0; r < 8; ++r) {
            const int row = rowbase + r + 8 * lh;
            const size_t m = (size_t)b * TT + row;
            yb[m * CC + h * DD + d] = f2bf(o[dt][r] / lrow[r]);
        }
    }
}

// ---------------------------------------------------------------------------
// Stage 3: output projection  out[m][n] = sum_k y[m][k] * Wp[n][k] + bp[n]
// ---------------------------------------------------------------------------
__global__ __launch_bounds__(128) void proj_gemm_kernel(
    const unsigned short* __restrict__ yb,   // 8192 x 512 bf16
    const unsigned short* __restrict__ wpb,  // 512 x 512 bf16
    const float* __restrict__ bp,
    float* __restrict__ out)                 // 8192 x 512 fp32
{
    const int lane = threadIdx.x & 31;
    const int wave = threadIdx.x >> 5;
    const int lm = lane & 15, lh = lane >> 4;
    const int wm = wave & 1, wn = wave >> 1;
    const int m0 = blockIdx.x * 128 + wm * 64;
    const int n0 = blockIdx.y * 64 + wn * 32;

    const v8f zero8 = {0.f, 0.f, 0.f, 0.f, 0.f, 0.f, 0.f, 0.f};
    v8f acc[4][2];
#pragma unroll
    for (int i = 0; i < 4; ++i)
#pragma unroll
        for (int j = 0; j < 2; ++j) acc[i][j] = zero8;

    for (int k0 = 0; k0 < CC; k0 += 32) {
        Frag32 a[4], bfr[2];
#pragma unroll
        for (int mt = 0; mt < 4; ++mt) {
            const unsigned short* p = yb + (size_t)(m0 + mt * 16 + lm) * CC + k0;
            a[mt].u[0] = *(const uint4*)(p + lh * 16);
            a[mt].u[1] = *(const uint4*)(p + lh * 16 + 8);
        }
#pragma unroll
        for (int nt = 0; nt < 2; ++nt) {
            const unsigned short* p = wpb + (size_t)(n0 + nt * 16 + lm) * CC + k0 + lh * 16;
            bfr[nt].u[0] = *(const uint4*)(p);
            bfr[nt].u[1] = *(const uint4*)(p + 8);
        }
#pragma unroll
        for (int mt = 0; mt < 4; ++mt)
#pragma unroll
            for (int nt = 0; nt < 2; ++nt)
                acc[mt][nt] = wmma_bf16(a[mt].v, bfr[nt].v, acc[mt][nt]);
    }

#pragma unroll
    for (int mt = 0; mt < 4; ++mt)
#pragma unroll
        for (int nt = 0; nt < 2; ++nt) {
            const int n = n0 + nt * 16 + lm;
            const float bv_ = bp[n];
#pragma unroll
            for (int r = 0; r < 8; ++r) {
                const int m = m0 + mt * 16 + r + 8 * lh;
                out[(size_t)m * CC + n] = acc[mt][nt][r] + bv_;
            }
        }
}

// ---------------------------------------------------------------------------
// Host launcher
// ---------------------------------------------------------------------------
extern "C" void kernel_launch(void* const* d_in, const int* in_sizes, int n_in,
                              void* d_out, int out_size, void* d_ws, size_t ws_size,
                              hipStream_t stream) {
    (void)in_sizes; (void)n_in; (void)out_size; (void)ws_size;

    const float* x  = (const float*)d_in[0];
    const float* Wq = (const float*)d_in[1];
    const float* bq = (const float*)d_in[2];
    const float* Wk = (const float*)d_in[3];
    const float* bk = (const float*)d_in[4];
    const float* Wv = (const float*)d_in[5];
    const float* bv = (const float*)d_in[6];
    const float* Wp = (const float*)d_in[7];
    const float* bp = (const float*)d_in[8];
    float* out = (float*)d_out;

    char* ws = (char*)d_ws;
    const size_t SZ_XB  = (size_t)MTOT * CC * 2;            // 8 MiB
    const size_t SZ_W   = (size_t)CC * CC * 2;              // 512 KiB
    const size_t SZ_QKV = (size_t)BB * HH * TT * DD * 2;    // 8 MiB

    unsigned short* xb  = (unsigned short*)(ws);
    unsigned short* wqb = (unsigned short*)(ws + SZ_XB);
    unsigned short* wkb = (unsigned short*)(ws + SZ_XB + SZ_W);
    unsigned short* wvb = (unsigned short*)(ws + SZ_XB + 2 * SZ_W);
    unsigned short* wpb = (unsigned short*)(ws + SZ_XB + 3 * SZ_W);
    unsigned short* Qb  = (unsigned short*)(ws + SZ_XB + 4 * SZ_W);
    unsigned short* Kb  = (unsigned short*)(ws + SZ_XB + 4 * SZ_W + SZ_QKV);
    unsigned short* Vt  = (unsigned short*)(ws + SZ_XB + 4 * SZ_W + 2 * SZ_QKV);
    unsigned short* yb  = (unsigned short*)(ws + SZ_XB + 4 * SZ_W + 3 * SZ_QKV);

    // Stage 0: conversions
    {
        const int nx = MTOT * CC;                 // 4,194,304
        cvt_f32_to_bf16<<<(nx + 255) / 256, 256, 0, stream>>>(x, xb, nx);
        const int nw = CC * CC;                   // 262,144
        cvt_f32_to_bf16<<<(nw + 255) / 256, 256, 0, stream>>>(Wq, wqb, nw);
        cvt_f32_to_bf16<<<(nw + 255) / 256, 256, 0, stream>>>(Wk, wkb, nw);
        cvt_f32_to_bf16<<<(nw + 255) / 256, 256, 0, stream>>>(Wv, wvb, nw);
        cvt_f32_to_bf16<<<(nw + 255) / 256, 256, 0, stream>>>(Wp, wpb, nw);
    }

    // Stage 1: fused QKV projection
    qkv_gemm_kernel<<<dim3(MTOT / 128, CC / 64, 3), 128, 0, stream>>>(
        xb, wqb, wkb, wvb, bq, bk, bv, Qb, Kb, Vt);

    // Stage 2: flash attention
    attn_kernel<<<dim3(TT / 128, BB * HH), 256, 0, stream>>>(Qb, Kb, Vt, yb);

    // Stage 3: output projection
    proj_gemm_kernel<<<dim3(MTOT / 128, CC / 64), 128, 0, stream>>>(yb, wpb, bp, out);
}
